// SlabCrossAttention3D2D_61211873903037
// MI455X (gfx1250) — compile-verified, matrix operand order checked by reading, not run on hardware
//
#include <hip/hip_runtime.h>

// ---------------------------------------------------------------------------
// Types for CDNA5 WMMA (wave32): V_WMMA_F32_16X16X32_BF16
// ---------------------------------------------------------------------------
typedef __attribute__((ext_vector_type(16))) __bf16 v16bf;
typedef __attribute__((ext_vector_type(8)))  __bf16 v8bf;
typedef __attribute__((ext_vector_type(8)))  float  v8f;

#define HWN   576      // 24*24
#define NEMB  256
#define NSLAB 8
#define ND    32

__device__ __forceinline__ __bf16 f2bf(float f) {
  unsigned u = __builtin_bit_cast(unsigned, f);
  unsigned r = u + 0x7FFFu + ((u >> 16) & 1u);   // round-to-nearest-even
  unsigned short hs = (unsigned short)(r >> 16);
  return __builtin_bit_cast(__bf16, hs);
}

// Load one lane's 16 bf16 elements for a 16x32 A tile (or 32x16 B tile,
// column-major source).  Per ISA layout, lane half h holds K = {h*8..h*8+7,
// 16+h*8..16+h*8+7} -> two contiguous 16-byte chunks.
__device__ __forceinline__ v16bf load_tile(const __bf16* p, int half) {
  const __bf16* q = p + half * 8;
  v8bf c0 = *(const v8bf*)(q);
  v8bf c1 = *(const v8bf*)(q + 16);
  v16bf r;
#pragma unroll
  for (int i = 0; i < 8; ++i) { r[i] = c0[i]; r[i + 8] = c1[i]; }
  return r;
}

__device__ __forceinline__ v8f wmma_bf16(v16bf a, v16bf b, v8f c) {
  return __builtin_amdgcn_wmma_f32_16x16x32_bf16(false, a, false, b,
                                                 (short)0, c, false, false);
}

// row reductions across the 16-lane half (C-layout: rows fixed per half)
__device__ __forceinline__ float rmax16(float v) {
#pragma unroll
  for (int m = 1; m < 16; m <<= 1) v = fmaxf(v, __shfl_xor(v, m, 32));
  return v;
}
__device__ __forceinline__ float rsum16(float v) {
#pragma unroll
  for (int m = 1; m < 16; m <<= 1) v += __shfl_xor(v, m, 32);
  return v;
}

// ---------------------------------------------------------------------------
// Pack / transpose kernels (f32 -> bf16)
// ---------------------------------------------------------------------------
__global__ void pack_x3_kernel(const float* __restrict__ F3, __bf16* __restrict__ x3h) {
  long i = (long)blockIdx.x * 256 + threadIdx.x;          // 18432*256
  if (i >= (long)ND * HWN * NEMB) return;
  int row = (int)(i >> 8), c = (int)(i & 255);
  x3h[i] = f2bf(F3[(size_t)c * (ND * HWN) + row]);
}

__global__ void pack_x2_kernel(const float* __restrict__ F2, __bf16* __restrict__ x2h) {
  long i = (long)blockIdx.x * 256 + threadIdx.x;          // 4608*128
  if (i >= (long)NSLAB * HWN * 128) return;
  int row = (int)(i >> 7), c = (int)(i & 127);
  int k = row / HWN, n = row - k * HWN;
  x2h[i] = f2bf(F2[(size_t)c * (HWN * NSLAB) + n * NSLAB + k]);
}

__global__ void packT_kernel(const float* __restrict__ src, __bf16* __restrict__ dst,
                             int rows, int cols) {
  long i = (long)blockIdx.x * 256 + threadIdx.x;          // dst[c*rows+r]=src[r*cols+c]
  if (i >= (long)rows * cols) return;
  int c = (int)(i / rows), r = (int)(i % rows);
  dst[i] = f2bf(src[(size_t)r * cols + c]);
}

// ---------------------------------------------------------------------------
// Topo conv branch (direct convs, VALU)
// ---------------------------------------------------------------------------
__global__ void conv1_kernel(const float* __restrict__ F2, const float* __restrict__ w,
                             const float* __restrict__ g, const float* __restrict__ b,
                             float* __restrict__ t1) {
  int i = blockIdx.x * 256 + threadIdx.x;                 // 8*64*576
  if (i >= NSLAB * 64 * HWN) return;
  int n = i % HWN, oc = (i / HWN) & 63, k = i / (HWN * 64);
  int y = n / 24, x = n % 24;
  float acc = 0.f;
  for (int ic = 0; ic < 128; ++ic) {
    const float* wp = w + (size_t)(oc * 128 + ic) * 9;
    const float* fp = F2 + (size_t)ic * (HWN * NSLAB) + k;
#pragma unroll
    for (int dy = 0; dy < 3; ++dy) {
      int yy = y + dy - 1; if (yy < 0 || yy >= 24) continue;
#pragma unroll
      for (int dx = 0; dx < 3; ++dx) {
        int xx = x + dx - 1; if (xx < 0 || xx >= 24) continue;
        acc += fp[(size_t)(yy * 24 + xx) * NSLAB] * wp[dy * 3 + dx];
      }
    }
  }
  float v = acc * (g[oc] * 0.9999950000374997f) + b[oc];  // BN eval, eps=1e-5
  t1[i] = fmaxf(v, 0.f);
}

__global__ void conv2_kernel(const float* __restrict__ t1, const float* __restrict__ w,
                             const float* __restrict__ g, const float* __restrict__ b,
                             float* __restrict__ t2) {
  int i = blockIdx.x * 256 + threadIdx.x;                 // 8*64*576
  if (i >= NSLAB * 64 * HWN) return;
  int n = i % HWN, oc = (i / HWN) & 63, k = i / (HWN * 64);
  int y = n / 24, x = n % 24;
  float acc = 0.f;
  for (int ic = 0; ic < 64; ++ic) {
    const float* wp = w + (size_t)(oc * 64 + ic) * 9;
    const float* fp = t1 + (size_t)(k * 64 + ic) * HWN;
#pragma unroll
    for (int dy = 0; dy < 3; ++dy) {
      int yy = y + dy - 1; if (yy < 0 || yy >= 24) continue;
#pragma unroll
      for (int dx = 0; dx < 3; ++dx) {
        int xx = x + dx - 1; if (xx < 0 || xx >= 24) continue;
        acc += fp[yy * 24 + xx] * wp[dy * 3 + dx];
      }
    }
  }
  float v = acc * (g[oc] * 0.9999950000374997f) + b[oc];
  t2[i] = fmaxf(v, 0.f);
}

__global__ void conv3_sig_kernel(const float* __restrict__ t2, const float* __restrict__ w3,
                                 const float* __restrict__ b3, float* __restrict__ p) {
  int i = blockIdx.x * 256 + threadIdx.x;                 // 8*576
  if (i >= NSLAB * HWN) return;
  int n = i % HWN, k = i / HWN;
  float acc = b3[0];
#pragma unroll 8
  for (int oc = 0; oc < 64; ++oc) acc += t2[(size_t)(k * 64 + oc) * HWN + n] * w3[oc];
  p[i] = 1.f / (1.f + __expf(-acc));
}

__global__ void topo_kernel(const float* __restrict__ p, float* __restrict__ topo) {
  int i = blockIdx.x * 256 + threadIdx.x;                 // 8*576
  if (i >= NSLAB * HWN) return;
  int n = i % HWN, k = i / HWN;
  int y = n / 24, x = n % 24;
  float s = 0.f;
#pragma unroll
  for (int dy = -1; dy <= 1; ++dy)
#pragma unroll
    for (int dx = -1; dx <= 1; ++dx) {
      if (dy == 0 && dx == 0) continue;
      int yy = y + dy, xx = x + dx;
      if (yy < 0 || yy >= 24 || xx < 0 || xx >= 24) continue;
      s += p[k * HWN + yy * 24 + xx];
    }
  topo[i] = p[i] + 0.5f * s;
}

// ---------------------------------------------------------------------------
// Fused GEMM (WMMA bf16) + LayerNorm.  Block = 256 threads (8 waves),
// computes 16 rows x 256 cols.  vmode: 0 -> out[(g4h)*576+n][d]  (Q/K layout)
//                                 1 -> out[(g4h)*64+d][n]        (V transposed)
// ---------------------------------------------------------------------------
__global__ void gemm_ln_kernel(const __bf16* __restrict__ X, int Kdim,
                               const __bf16* __restrict__ WT,
                               const float* __restrict__ g, const float* __restrict__ b,
                               __bf16* __restrict__ out, int vmode) {
  __shared__ float buf[16][NEMB];
  __shared__ float red[16][16][2];
  __shared__ float stats[16][2];
  int tid = threadIdx.x;
  int wv = tid >> 5, lane = tid & 31, half = lane >> 4, idx = lane & 15;
  int R0 = blockIdx.x * 16;

  v8f c0 = {0.f,0.f,0.f,0.f,0.f,0.f,0.f,0.f};
  v8f c1 = c0;
  for (int k0 = 0; k0 < Kdim; k0 += 32) {
    v16bf a  = load_tile(X  + (size_t)(R0 + idx) * Kdim + k0, half);
    v16bf b0 = load_tile(WT + (size_t)(wv * 32 + idx)      * Kdim + k0, half);
    v16bf b1 = load_tile(WT + (size_t)(wv * 32 + 16 + idx) * Kdim + k0, half);
    c0 = wmma_bf16(a, b0, c0);
    c1 = wmma_bf16(a, b1, c1);
  }
#pragma unroll
  for (int v = 0; v < 8; ++v) {
    buf[v + 8 * half][wv * 32 + idx]      = c0[v];
    buf[v + 8 * half][wv * 32 + 16 + idx] = c1[v];
  }
  __syncthreads();
  {
    int r = tid >> 4, seg = tid & 15;
    float s = 0.f, s2 = 0.f;
#pragma unroll
    for (int i = 0; i < 16; ++i) { float x = buf[r][seg * 16 + i]; s += x; s2 += x * x; }
    red[r][seg][0] = s; red[r][seg][1] = s2;
  }
  __syncthreads();
  if (tid < 16) {
    float s = 0.f, s2 = 0.f;
#pragma unroll
    for (int i = 0; i < 16; ++i) { s += red[tid][i][0]; s2 += red[tid][i][1]; }
    float mu = s * (1.f / NEMB);
    float var = s2 * (1.f / NEMB) - mu * mu;
    stats[tid][0] = mu; stats[tid][1] = rsqrtf(var + 1e-5f);
  }
  __syncthreads();
  {
    int r = tid >> 4, seg = tid & 15;
    float mu = stats[r][0], rs = stats[r][1];
    int row = R0 + r, grp = row / HWN, n = row - grp * HWN;
#pragma unroll
    for (int i = 0; i < 16; ++i) {
      int c = seg * 16 + i;
      float y = (buf[r][c] - mu) * rs * g[c] + b[c];
      int h = c >> 6, d = c & 63;
      size_t o = (vmode == 0)
        ? ((size_t)((grp * 4 + h) * HWN + n) * 64 + d)
        : ((size_t)((grp * 4 + h) * 64 + d) * HWN + n);
      out[o] = f2bf(y);
    }
  }
}

// ---------------------------------------------------------------------------
// Flash attention over 3-slab window.  1 wave per block, 16 Q-rows per block.
// grid = (36 row tiles, 4 heads, 32 depth slices), block = 32.
// ---------------------------------------------------------------------------
__global__ void attn_kernel(const __bf16* __restrict__ Qh, const __bf16* __restrict__ Kh,
                            const __bf16* __restrict__ Vt, const float* __restrict__ topo,
                            const float* __restrict__ lamp, const int* __restrict__ D0p,
                            const int* __restrict__ slabp, __bf16* __restrict__ Oh) {
  __shared__ __bf16 lds_p[16][32];
  int lane = threadIdx.x, half = lane >> 4, idx = lane & 15;
  int R0 = blockIdx.x * 16;
  int h  = blockIdx.y;
  int z  = blockIdx.z;
  int D0 = *D0p, slab = *slabp;
  float lam = *lamp;
  int z0 = (int)rintf((float)z * (float)(D0 - 1) / 31.0f);   // jnp.round(linspace)
  int kidx = z0 / slab; kidx = min(max(kidx, 0), NSLAB - 1);

  const __bf16* Qbase = Qh + ((size_t)(z * 4 + h) * HWN + R0) * 64;
  v16bf qa0 = load_tile(Qbase + idx * 64 + 0,  half);
  v16bf qa1 = load_tile(Qbase + idx * 64 + 32, half);

  v8f o0 = {0.f,0.f,0.f,0.f,0.f,0.f,0.f,0.f}, o1 = o0, o2 = o0, o3 = o0;
  float M[8], L[8];
#pragma unroll
  for (int v = 0; v < 8; ++v) { M[v] = -3.0e38f; L[v] = 0.f; }

  for (int j = 0; j < 3; ++j) {
    int s = kidx + j - 1;
    if (s < 0 || s >= NSLAB) continue;                      // invalid window -> exp(NEG)=0
    float bias = (j == 1) ? 0.f : -0.5f;                    // slab bias
    const __bf16* Kbase = Kh + (size_t)(s * 4 + h) * HWN * 64;
    const __bf16* Vbase = Vt + (size_t)(s * 4 + h) * 64 * HWN;
    const float*  tp    = topo + s * HWN;

    for (int m0 = 0; m0 < HWN; m0 += 32) {
      if (m0 + 32 < HWN)
        __builtin_prefetch(Kbase + (size_t)(m0 + 32 + idx) * 64, 0, 0);

      v16bf kb0a = load_tile(Kbase + (size_t)(m0 + idx) * 64 + 0,       half);
      v16bf kb0b = load_tile(Kbase + (size_t)(m0 + idx) * 64 + 32,      half);
      v16bf kb1a = load_tile(Kbase + (size_t)(m0 + 16 + idx) * 64 + 0,  half);
      v16bf kb1b = load_tile(Kbase + (size_t)(m0 + 16 + idx) * 64 + 32, half);
      v8f s0 = {0.f,0.f,0.f,0.f,0.f,0.f,0.f,0.f}, s1 = s0;
      s0 = wmma_bf16(qa0, kb0a, s0);  s0 = wmma_bf16(qa1, kb0b, s0);
      s1 = wmma_bf16(qa0, kb1a, s1);  s1 = wmma_bf16(qa1, kb1b, s1);

      float p0[8], p1[8], sc[8];
#pragma unroll
      for (int v = 0; v < 8; ++v) {
        int row = R0 + v + 8 * half;
        float a0 = s0[v] * 0.125f + bias;                   // 1/sqrt(64)
        float a1 = s1[v] * 0.125f + bias;
        if (row == m0 + idx)      a0 += lam * tp[row];      // diagonal topo
        if (row == m0 + 16 + idx) a1 += lam * tp[row];
        float t = rmax16(fmaxf(a0, a1));
        float mn = fmaxf(M[v], t);
        sc[v] = __expf(M[v] - mn);
        p0[v] = __expf(a0 - mn);
        p1[v] = __expf(a1 - mn);
        L[v] = L[v] * sc[v] + rsum16(p0[v] + p1[v]);
        M[v] = mn;
      }
#pragma unroll
      for (int v = 0; v < 8; ++v) {
        lds_p[v + 8 * half][idx]      = f2bf(p0[v]);
        lds_p[v + 8 * half][16 + idx] = f2bf(p1[v]);
      }
      __syncthreads();
      v16bf pa = load_tile(&lds_p[idx][0], half);           // C-layout -> A-layout
#pragma unroll
      for (int v = 0; v < 8; ++v) {
        o0[v] *= sc[v]; o1[v] *= sc[v]; o2[v] *= sc[v]; o3[v] *= sc[v];
      }
      v16bf vb0 = load_tile(Vbase + (size_t)(0 * 16 + idx) * HWN + m0, half);
      v16bf vb1 = load_tile(Vbase + (size_t)(1 * 16 + idx) * HWN + m0, half);
      v16bf vb2 = load_tile(Vbase + (size_t)(2 * 16 + idx) * HWN + m0, half);
      v16bf vb3 = load_tile(Vbase + (size_t)(3 * 16 + idx) * HWN + m0, half);
      o0 = wmma_bf16(pa, vb0, o0);
      o1 = wmma_bf16(pa, vb1, o1);
      o2 = wmma_bf16(pa, vb2, o2);
      o3 = wmma_bf16(pa, vb3, o3);
      __syncthreads();
    }
  }
#pragma unroll
  for (int v = 0; v < 8; ++v) {
    float inv = 1.0f / L[v];
    size_t base = ((size_t)(z * HWN + R0 + v + 8 * half)) * NEMB + h * 64;
    Oh[base + idx]      = f2bf(o0[v] * inv);
    Oh[base + 16 + idx] = f2bf(o1[v] * inv);
    Oh[base + 32 + idx] = f2bf(o2[v] * inv);
    Oh[base + 48 + idx] = f2bf(o3[v] * inv);
  }
}

// ---------------------------------------------------------------------------
// Output projection: out = F3 + (O @ Wproj) scattered to NCDHW
// ---------------------------------------------------------------------------
__global__ void proj_kernel(const __bf16* __restrict__ Oh, const __bf16* __restrict__ WpT,
                            const float* __restrict__ F3, float* __restrict__ out) {
  int tid = threadIdx.x;
  int wv = tid >> 5, lane = tid & 31, half = lane >> 4, idx = lane & 15;
  int R0 = blockIdx.x * 16;
  v8f c0 = {0.f,0.f,0.f,0.f,0.f,0.f,0.f,0.f};
  v8f c1 = c0;
  for (int k0 = 0; k0 < NEMB; k0 += 32) {
    v16bf a  = load_tile(Oh  + (size_t)(R0 + idx) * NEMB + k0, half);
    v16bf b0 = load_tile(WpT + (size_t)(wv * 32 + idx)      * NEMB + k0, half);
    v16bf b1 = load_tile(WpT + (size_t)(wv * 32 + 16 + idx) * NEMB + k0, half);
    c0 = wmma_bf16(a, b0, c0);
    c1 = wmma_bf16(a, b1, c1);
  }
#pragma unroll
  for (int v = 0; v < 8; ++v) {
    int row = R0 + v + 8 * half;
    int cA = wv * 32 + idx, cB = wv * 32 + 16 + idx;
    out[(size_t)cA * (ND * HWN) + row] = F3[(size_t)cA * (ND * HWN) + row] + c0[v];
    out[(size_t)cB * (ND * HWN) + row] = F3[(size_t)cB * (ND * HWN) + row] + c1[v];
  }
}

// ---------------------------------------------------------------------------
// Host launcher
// ---------------------------------------------------------------------------
extern "C" void kernel_launch(void* const* d_in, const int* in_sizes, int n_in,
                              void* d_out, int out_size, void* d_ws, size_t ws_size,
                              hipStream_t stream) {
  (void)in_sizes; (void)n_in; (void)out_size; (void)ws_size;
  const float* F3     = (const float*)d_in[0];
  const float* F2     = (const float*)d_in[1];
  const int*   D0p    = (const int*)d_in[2];
  const int*   slabp  = (const int*)d_in[3];
  const float* Wq     = (const float*)d_in[4];
  const float* Wk     = (const float*)d_in[5];
  const float* Wv     = (const float*)d_in[6];
  const float* Wproj  = (const float*)d_in[7];
  const float* lnq_g  = (const float*)d_in[8];
  const float* lnq_b  = (const float*)d_in[9];
  const float* lnkv_g = (const float*)d_in[10];
  const float* lnkv_b = (const float*)d_in[11];
  const float* c1w    = (const float*)d_in[12];
  const float* bn1g   = (const float*)d_in[13];
  const float* bn1b   = (const float*)d_in[14];
  const float* c2w    = (const float*)d_in[15];
  const float* bn2g   = (const float*)d_in[16];
  const float* bn2b   = (const float*)d_in[17];
  const float* c3w    = (const float*)d_in[18];
  const float* c3b    = (const float*)d_in[19];
  const float* lam    = (const float*)d_in[20];
  float* out = (float*)d_out;

  char* w = (char*)d_ws;
  auto alloc = [&](size_t bytes) -> void* {
    void* p = (void*)w;
    w += (bytes + 255) & ~(size_t)255;
    return p;
  };
  const size_t MQ = (size_t)ND * HWN;      // 18432
  const size_t MK = (size_t)NSLAB * HWN;   // 4608
  __bf16* x3h = (__bf16*)alloc(MQ * 256 * 2);
  __bf16* x2h = (__bf16*)alloc(MK * 128 * 2);
  __bf16* WqT = (__bf16*)alloc(256 * 256 * 2);
  __bf16* WkT = (__bf16*)alloc(256 * 128 * 2);
  __bf16* WvT = (__bf16*)alloc(256 * 128 * 2);
  __bf16* WpT = (__bf16*)alloc(256 * 256 * 2);
  __bf16* Qh  = (__bf16*)alloc(MQ * 256 * 2);
  __bf16* Kh  = (__bf16*)alloc(MK * 256 * 2);
  __bf16* Vt  = (__bf16*)alloc(MK * 256 * 2);
  __bf16* Oh  = (__bf16*)alloc(MQ * 256 * 2);
  float*  t1  = (float*)alloc((size_t)NSLAB * 64 * HWN * 4);
  float*  t2  = (float*)alloc((size_t)NSLAB * 64 * HWN * 4);
  float*  pp  = (float*)alloc((size_t)NSLAB * HWN * 4);
  float*  tp  = (float*)alloc((size_t)NSLAB * HWN * 4);

  auto cdiv = [](long a, long b) { return (int)((a + b - 1) / b); };

  // pack
  pack_x3_kernel<<<cdiv(MQ * 256, 256), 256, 0, stream>>>(F3, x3h);
  pack_x2_kernel<<<cdiv(MK * 128, 256), 256, 0, stream>>>(F2, x2h);
  packT_kernel<<<cdiv(256 * 256, 256), 256, 0, stream>>>(Wq, WqT, 256, 256);
  packT_kernel<<<cdiv(128 * 256, 256), 256, 0, stream>>>(Wk, WkT, 128, 256);
  packT_kernel<<<cdiv(128 * 256, 256), 256, 0, stream>>>(Wv, WvT, 128, 256);
  packT_kernel<<<cdiv(256 * 256, 256), 256, 0, stream>>>(Wproj, WpT, 256, 256);

  // topo branch
  conv1_kernel<<<cdiv((long)NSLAB * 64 * HWN, 256), 256, 0, stream>>>(F2, c1w, bn1g, bn1b, t1);
  conv2_kernel<<<cdiv((long)NSLAB * 64 * HWN, 256), 256, 0, stream>>>(t1, c2w, bn2g, bn2b, t2);
  conv3_sig_kernel<<<cdiv((long)NSLAB * HWN, 256), 256, 0, stream>>>(t2, c3w, c3b, pp);
  topo_kernel<<<cdiv((long)NSLAB * HWN, 256), 256, 0, stream>>>(pp, tp);

  // Q / K / V projections + layernorm (WMMA)
  gemm_ln_kernel<<<(int)(MQ / 16), 256, 0, stream>>>(x3h, 256, WqT, lnq_g, lnq_b, Qh, 0);
  gemm_ln_kernel<<<(int)(MK / 16), 256, 0, stream>>>(x2h, 128, WkT, lnkv_g, lnkv_b, Kh, 0);
  gemm_ln_kernel<<<(int)(MK / 16), 256, 0, stream>>>(x2h, 128, WvT, lnkv_g, lnkv_b, Vt, 1);

  // flash attention over slab window (WMMA)
  attn_kernel<<<dim3(HWN / 16, 4, ND), 32, 0, stream>>>(Qh, Kh, Vt, tp, lam, D0p, slabp, Oh);

  // output projection + residual (WMMA)
  proj_kernel<<<(int)(MQ / 16), 256, 0, stream>>>(Oh, WpT, F3, out);
}
